// Block_23811298689439
// MI455X (gfx1250) — compile-verified, matrix operand order checked by reading, not run on hardware
//
#include <hip/hip_runtime.h>
#include <hip/hip_bf16.h>
#include <math.h>

typedef __attribute__((ext_vector_type(16))) __bf16 v16bf;
typedef __attribute__((ext_vector_type(8)))  float  v8f;

// ---------------------------------------------------------------------------
// WMMA helpers (CDNA5 wave32, 16x16x32 bf16 -> f32)
// ---------------------------------------------------------------------------
__device__ __forceinline__ v8f wmma_bf16(v16bf a, v16bf b, v8f c) {
  // 8 args: (neg_a, A, neg_b, B, c_mod, C, reuse_a, reuse_b)
  return __builtin_amdgcn_wmma_f32_16x16x32_bf16(false, a, false, b, (short)0, c,
                                                 false, false);
}

// Load a 16x32 bf16 fragment from a row-major matrix (works for A, and for B
// when given B^T stored row-major N x K).  ISA A-layout: lane holds row
// (lane&15); lanes 0-15 hold K {0..7, 16..23}, lanes 16-31 hold K {8..15,
// 24..31} -> two contiguous 16B chunks per lane.
__device__ __forceinline__ v16bf frag_rm(const __bf16* base, int row0, int k0,
                                         int ld, int lane) {
  int m16 = lane & 15, lh = lane >> 4;
  const __bf16* p = base + (size_t)(row0 + m16) * ld + k0 + (lh << 3);
  union { v16bf v; uint4 u[2]; } w;
  w.u[0] = *(const uint4*)p;
  w.u[1] = *(const uint4*)(p + 16);
  return w.v;
}

// Same, but rows clamped to <=195 (window token rows), fixed ld = 3072.
__device__ __forceinline__ v16bf frag_clamp196(const __bf16* base, int row0,
                                               int k0, int lane) {
  int m16 = lane & 15, lh = lane >> 4;
  int r = row0 + m16;
  r = (r > 195) ? 195 : r;
  const __bf16* p = base + (size_t)r * 3072 + k0 + (lh << 3);
  union { v16bf v; uint4 u[2]; } w;
  w.u[0] = *(const uint4*)p;
  w.u[1] = *(const uint4*)(p + 16);
  return w.v;
}

// ---------------------------------------------------------------------------
// Weight convert: W (K x N, f32) -> Wt (N x K, bf16), LDS tile transpose
// ---------------------------------------------------------------------------
__global__ __launch_bounds__(256) void wconv_kernel(const float* __restrict__ W,
                                                    __bf16* __restrict__ Wt,
                                                    int K, int N) {
  __shared__ float tile[32][33];
  int k0 = blockIdx.x * 32, n0 = blockIdx.y * 32;
  int tx = threadIdx.x & 31, ty = threadIdx.x >> 5;   // 32 x 8
  for (int r = ty; r < 32; r += 8)
    tile[r][tx] = W[(size_t)(k0 + r) * N + n0 + tx];
  __syncthreads();
  for (int r = ty; r < 32; r += 8)
    Wt[(size_t)(n0 + r) * K + k0 + tx] = (__bf16)tile[tx][r];
}

// ---------------------------------------------------------------------------
// LayerNorm (DIM=1024).  WIN=true: also window-partition (pad rows -> zeros,
// matching reference which pads AFTER the norm).  Output bf16.
// ---------------------------------------------------------------------------
template <bool WIN>
__global__ __launch_bounds__(256) void ln_kernel(const float* __restrict__ x,
                                                 const float* __restrict__ scale,
                                                 const float* __restrict__ bias,
                                                 __bf16* __restrict__ out) {
  int m = blockIdx.x;
  int srcRow = -1;
  if (WIN) {
    if (m < 19600) {
      int wi = m / 196, t = m % 196;
      int b = wi / 25, r2 = wi % 25, bi = r2 / 5, bj = r2 % 5;
      int hh = bi * 14 + t / 14, ww = bj * 14 + t % 14;
      if (hh < 64 && ww < 64) srcRow = (b * 64 + hh) * 64 + ww;
    }
  } else {
    srcRow = m;
  }
  float4 v = {0.f, 0.f, 0.f, 0.f};
  if (srcRow >= 0)
    v = ((const float4*)(x + (size_t)srcRow * 1024))[threadIdx.x];
  float s = v.x + v.y + v.z + v.w;
  float s2 = v.x * v.x + v.y * v.y + v.z * v.z + v.w * v.w;
#pragma unroll
  for (int off = 1; off < 32; off <<= 1) {
    s += __shfl_xor(s, off, 32);
    s2 += __shfl_xor(s2, off, 32);
  }
  __shared__ float red0[8], red1[8];
  int wave = threadIdx.x >> 5, lane = threadIdx.x & 31;
  if (lane == 0) { red0[wave] = s; red1[wave] = s2; }
  __syncthreads();
  s = 0.f; s2 = 0.f;
#pragma unroll
  for (int i = 0; i < 8; ++i) { s += red0[i]; s2 += red1[i]; }
  float mean = s * (1.f / 1024.f);
  float var = s2 * (1.f / 1024.f) - mean * mean;
  float rstd = rsqrtf(var + 1e-6f);
  int c0 = threadIdx.x * 4;
  union { __bf16 b[4]; uint2 u; } o;
  float vv[4] = {v.x, v.y, v.z, v.w};
#pragma unroll
  for (int j = 0; j < 4; ++j) {
    float r = (srcRow >= 0) ? (vv[j] - mean) * rstd * scale[c0 + j] + bias[c0 + j]
                            : 0.f;
    o.b[j] = (__bf16)r;
  }
  *(uint2*)(out + (size_t)m * 1024 + c0) = o.u;
}

// ---------------------------------------------------------------------------
// Generic WMMA GEMM: C[M,N] = act(A[M,K] @ Bt[N,K]^T + bias) (+ residual)
// Block tile 128x128, K-step 64, 8 waves (4x2), wave tile 32x64 (2x4 WMMA
// tiles).  Register-staged double buffer: next K-slab is loaded to VGPRs
// while the current LDS slab feeds the WMMA chain; k+2 slab prefetched to
// cache via global_prefetch.
// MODE 0: bf16 store                                (QKV)
// MODE 1: window-unpartition + residual, f32 store  (proj)
// MODE 2: exact GELU, bf16 store                    (fc1)
// MODE 3: residual, f32 store                       (fc2 -> d_out)
// ---------------------------------------------------------------------------
template <int MODE>
__global__ __launch_bounds__(256) void gemm_kernel(
    const __bf16* __restrict__ A, const __bf16* __restrict__ Bt,
    const float* __restrict__ bias, void* __restrict__ Cout,
    const float* __restrict__ resid, int N, int K) {
  __shared__ __bf16 As[128][72];   // +8 pad: conflict-free b128 frag reads
  __shared__ __bf16 Bs[128][72];
  int m0 = blockIdx.x * 128;
  int n0 = blockIdx.y * 128;
  int tid = threadIdx.x;
  int wave = tid >> 5, lane = tid & 31;
  int wm = wave >> 1, wn = wave & 1;       // 4 x 2 wave grid
  int m16 = lane & 15, lh = lane >> 4;

  const __bf16* Ag = A + (size_t)m0 * K;
  const __bf16* Bg = Bt + (size_t)n0 * K;
  int r = tid >> 3, cc = tid & 7;          // 32 rows x 8 chunks per pass

  uint4 aReg[4], bReg[4];
#pragma unroll
  for (int i = 0; i < 4; ++i) {
    aReg[i] = *(const uint4*)&Ag[(size_t)(r + i * 32) * K + cc * 8];
    bReg[i] = *(const uint4*)&Bg[(size_t)(r + i * 32) * K + cc * 8];
  }

  v8f acc[2][4] = {};
  for (int k0 = 0; k0 < K; k0 += 64) {
#pragma unroll
    for (int i = 0; i < 4; ++i) {
      *(uint4*)&As[r + i * 32][cc * 8] = aReg[i];
      *(uint4*)&Bs[r + i * 32][cc * 8] = bReg[i];
    }
    __syncthreads();
    int k1 = k0 + 64;
    if (k1 < K) {
#pragma unroll
      for (int i = 0; i < 4; ++i) {
        aReg[i] = *(const uint4*)&Ag[(size_t)(r + i * 32) * K + k1 + cc * 8];
        bReg[i] = *(const uint4*)&Bg[(size_t)(r + i * 32) * K + k1 + cc * 8];
      }
    }
    if (k0 + 128 < K) {  // warm L2/WGP$ for the slab after next
      __builtin_prefetch(&Ag[(size_t)r * K + k0 + 128 + cc * 8], 0, 1);
      __builtin_prefetch(&Bg[(size_t)r * K + k0 + 128 + cc * 8], 0, 1);
    }
#pragma unroll
    for (int ks = 0; ks < 64; ks += 32) {
      v16bf af[2], bf[4];
#pragma unroll
      for (int i = 0; i < 2; ++i)
        af[i] = frag_rm(&As[0][0], wm * 32 + i * 16, ks, 72, lane);
#pragma unroll
      for (int j = 0; j < 4; ++j)
        bf[j] = frag_rm(&Bs[0][0], wn * 64 + j * 16, ks, 72, lane);
#pragma unroll
      for (int i = 0; i < 2; ++i)
#pragma unroll
        for (int j = 0; j < 4; ++j)
          acc[i][j] = wmma_bf16(af[i], bf[j], acc[i][j]);
    }
    __syncthreads();
  }

#pragma unroll
  for (int i2 = 0; i2 < 2; ++i2)
#pragma unroll
    for (int j2 = 0; j2 < 4; ++j2)
#pragma unroll
      for (int vi = 0; vi < 8; ++vi) {
        int row = m0 + wm * 32 + i2 * 16 + vi + 8 * lh;
        int col = n0 + wn * 64 + j2 * 16 + m16;
        float c = acc[i2][j2][vi] + bias[col];
        if (MODE == 0) {
          ((__bf16*)Cout)[(size_t)row * N + col] = (__bf16)c;
        } else if (MODE == 1) {
          if (row < 19600) {
            int wi = row / 196, t = row % 196;
            int b = wi / 25, r2 = wi % 25, bi = r2 / 5, bj = r2 % 5;
            int hh = bi * 14 + t / 14, ww = bj * 14 + t % 14;
            if (hh < 64 && ww < 64) {
              size_t idx = (((size_t)b * 64 + hh) * 64 + ww) * 1024 + col;
              ((float*)Cout)[idx] = resid[idx] + c;
            }
          }
        } else if (MODE == 2) {
          float g = 0.5f * c * (1.0f + erff(c * 0.70710678118654752f));
          ((__bf16*)Cout)[(size_t)row * N + col] = (__bf16)g;
        } else {
          size_t idx = (size_t)row * N + col;
          ((float*)Cout)[idx] = resid[idx] + c;
        }
      }
}

// ---------------------------------------------------------------------------
// Windowed attention with decomposed rel-pos.  One block per (window, head).
// S kept in registers (13 tiles/wave/query-tile), softmax via lane shuffles,
// P transposed through wave-private LDS for the P@V WMMAs.
// ---------------------------------------------------------------------------
__global__ __launch_bounds__(256) void attn_kernel(
    const __bf16* __restrict__ qkv, const float* __restrict__ rph,
    const float* __restrict__ rpw, __bf16* __restrict__ attn_out) {
  __shared__ float RH[196][16];        // 12.25 KB
  __shared__ float RW[196][16];        // 12.25 KB
  __shared__ __bf16 vT[64][216];       // v^T (d, key) + pad, 27 KB
  __shared__ __bf16 Pscr[8][16][40];   // wave-private P chunk, 10 KB
  int head = blockIdx.x;
  int wbase = blockIdx.y * 196;
  int tid = threadIdx.x;
  const __bf16* qb = qkv + (size_t)wbase * 3072 + head * 64;
  const __bf16* kb = qb + 1024;
  const __bf16* vb = qb + 2048;

  // rel-pos dot tables: RH[t][ki] = q[t,:].rel_pos_h[qi-ki+13,:], same for RW
  for (int e = tid; e < 196 * 28; e += 256) {
    int t = e / 28, rr = e % 28, tab = rr / 14, kk = rr % 14;
    int qi = t / 14, qj = t % 14;
    const float* rp = (tab ? rpw : rph) + ((tab ? qj : qi) - kk + 13) * 64;
    const uint4* q4 = (const uint4*)(qb + (size_t)t * 3072);
    float s = 0.f;
#pragma unroll
    for (int d8 = 0; d8 < 8; ++d8) {
      union { uint4 u; __bf16 b[8]; } w;
      w.u = q4[d8];
      float4 r0 = ((const float4*)(rp + d8 * 8))[0];
      float4 r1 = ((const float4*)(rp + d8 * 8))[1];
      s += (float)w.b[0] * r0.x + (float)w.b[1] * r0.y + (float)w.b[2] * r0.z +
           (float)w.b[3] * r0.w + (float)w.b[4] * r1.x + (float)w.b[5] * r1.y +
           (float)w.b[6] * r1.z + (float)w.b[7] * r1.w;
    }
    if (tab) RW[t][kk] = s; else RH[t][kk] = s;
  }
  // v^T staging (keys >=196 zeroed)
  for (int e = tid; e < 64 * 208; e += 256) {
    int d = e / 208, key = e % 208;
    vT[d][key] = (key < 196) ? vb[(size_t)key * 3072 + d] : (__bf16)0.f;
  }
  __syncthreads();

  int wave = tid >> 5, lane = tid & 31;
  int m16 = lane & 15, lh = lane >> 4;

  for (int qt = wave; qt < 13; qt += 8) {
    v16bf qf0 = frag_clamp196(qb, qt * 16, 0, lane);
    v16bf qf1 = frag_clamp196(qb, qt * 16, 32, lane);
    v8f sacc[13];
#pragma unroll
    for (int kt = 0; kt < 13; ++kt) {
      v16bf kf0 = frag_clamp196(kb, kt * 16, 0, lane);
      v16bf kf1 = frag_clamp196(kb, kt * 16, 32, lane);
      v8f a = {};
      a = wmma_bf16(qf0, kf0, a);
      a = wmma_bf16(qf1, kf1, a);
      sacc[kt] = a;
    }
    // scale + rel-pos + mask, track row max
    float mx[8];
#pragma unroll
    for (int vi = 0; vi < 8; ++vi) mx[vi] = -3.0e38f;
#pragma unroll
    for (int kt = 0; kt < 13; ++kt) {
      int key = kt * 16 + m16;
      int ki = key / 14, kj = key % 14;
      bool kok = (key < 196);
#pragma unroll
      for (int vi = 0; vi < 8; ++vi) {
        int qrow = qt * 16 + vi + 8 * lh;
        float s;
        if (kok && qrow < 196)
          s = sacc[kt][vi] * 0.125f + RH[qrow][ki] + RW[qrow][kj];
        else
          s = -3.0e38f;
        sacc[kt][vi] = s;
        mx[vi] = fmaxf(mx[vi], s);
      }
    }
    // softmax: reduce over the 16-lane half holding this row group
#pragma unroll
    for (int vi = 0; vi < 8; ++vi)
#pragma unroll
      for (int sft = 1; sft < 16; sft <<= 1)
        mx[vi] = fmaxf(mx[vi], __shfl_xor(mx[vi], sft, 32));
    float sum[8];
#pragma unroll
    for (int vi = 0; vi < 8; ++vi) sum[vi] = 0.f;
#pragma unroll
    for (int kt = 0; kt < 13; ++kt)
#pragma unroll
      for (int vi = 0; vi < 8; ++vi) {
        float p = __expf(sacc[kt][vi] - mx[vi]);
        sacc[kt][vi] = p;
        sum[vi] += p;
      }
#pragma unroll
    for (int vi = 0; vi < 8; ++vi) {
#pragma unroll
      for (int sft = 1; sft < 16; sft <<= 1)
        sum[vi] += __shfl_xor(sum[vi], sft, 32);
      sum[vi] = 1.0f / sum[vi];
    }
#pragma unroll
    for (int kt = 0; kt < 13; ++kt)
#pragma unroll
      for (int vi = 0; vi < 8; ++vi) sacc[kt][vi] *= sum[vi];

    // P @ V over key chunks of 32 (transpose P via wave-private LDS)
    v8f oacc[4] = {};
    __bf16 (*Pw)[40] = Pscr[wave];
    for (int c = 0; c < 7; ++c) {
#pragma unroll
      for (int half = 0; half < 2; ++half) {
        int kt = c * 2 + half;
#pragma unroll
        for (int vi = 0; vi < 8; ++vi) {
          float p = (kt < 13) ? sacc[kt][vi] : 0.f;
          Pw[vi + 8 * lh][half * 16 + m16] = (__bf16)p;
        }
      }
      asm volatile("" ::: "memory");  // LDS in-order per wave; block reorder
      v16bf pf = frag_rm(&Pw[0][0], 0, 0, 40, lane);
      asm volatile("" ::: "memory");
#pragma unroll
      for (int nt = 0; nt < 4; ++nt) {
        v16bf vf = frag_rm(&vT[0][0], nt * 16, c * 32, 216, lane);
        oacc[nt] = wmma_bf16(pf, vf, oacc[nt]);
      }
    }
    // store (token, head*64 + d) layout for the proj GEMM
#pragma unroll
    for (int nt = 0; nt < 4; ++nt)
#pragma unroll
      for (int vi = 0; vi < 8; ++vi) {
        int row = qt * 16 + vi + 8 * lh;
        if (row < 196)
          attn_out[(size_t)(wbase + row) * 1024 + head * 64 + nt * 16 + m16] =
              (__bf16)oacc[nt][vi];
      }
  }
}

// ---------------------------------------------------------------------------
// Host orchestration
// ---------------------------------------------------------------------------
extern "C" void kernel_launch(void* const* d_in, const int* in_sizes, int n_in,
                              void* d_out, int out_size, void* d_ws,
                              size_t ws_size, hipStream_t stream) {
  const float* x     = (const float*)d_in[0];
  const float* n1s   = (const float*)d_in[1];
  const float* n1b   = (const float*)d_in[2];
  const float* qkvW  = (const float*)d_in[3];
  const float* qkvB  = (const float*)d_in[4];
  const float* rph   = (const float*)d_in[5];
  const float* rpw   = (const float*)d_in[6];
  const float* projW = (const float*)d_in[7];
  const float* projB = (const float*)d_in[8];
  const float* n2s   = (const float*)d_in[9];
  const float* n2b   = (const float*)d_in[10];
  const float* fc1W  = (const float*)d_in[11];
  const float* fc1B  = (const float*)d_in[12];
  const float* fc2W  = (const float*)d_in[13];
  const float* fc2B  = (const float*)d_in[14];

  char* ws = (char*)d_ws;
  size_t off = 0;
  auto take = [&](size_t bytes) -> void* {
    void* p = ws + off;
    off = (off + bytes + 255) & ~(size_t)255;
    return p;
  };
  const size_t MP = 19712;  // 19600 window tokens padded to 154*128
  __bf16* xw      = (__bf16*)take(MP * 1024 * 2);
  __bf16* WqkvT   = (__bf16*)take((size_t)3072 * 1024 * 2);
  __bf16* WprojT  = (__bf16*)take((size_t)1024 * 1024 * 2);
  __bf16* Wfc1T   = (__bf16*)take((size_t)4096 * 1024 * 2);
  __bf16* Wfc2T   = (__bf16*)take((size_t)1024 * 4096 * 2);
  __bf16* qkvact  = (__bf16*)take(MP * 3072 * 2);
  __bf16* attnout = (__bf16*)take(MP * 1024 * 2);
  float*  x2      = (float*)take((size_t)16384 * 1024 * 4);
  __bf16* z1      = (__bf16*)take((size_t)16384 * 1024 * 2);
  __bf16* h1      = (__bf16*)take((size_t)16384 * 4096 * 2);

  // 1) weights -> transposed bf16
  wconv_kernel<<<dim3(32, 96), 256, 0, stream>>>(qkvW, WqkvT, 1024, 3072);
  wconv_kernel<<<dim3(32, 32), 256, 0, stream>>>(projW, WprojT, 1024, 1024);
  wconv_kernel<<<dim3(32, 128), 256, 0, stream>>>(fc1W, Wfc1T, 1024, 4096);
  wconv_kernel<<<dim3(128, 32), 256, 0, stream>>>(fc2W, Wfc2T, 4096, 1024);
  // 2) LN1 + window partition
  ln_kernel<true><<<dim3(19712), 256, 0, stream>>>(x, n1s, n1b, xw);
  // 3) QKV GEMM (19712 x 3072 x 1024)
  gemm_kernel<0><<<dim3(154, 24), 256, 0, stream>>>(xw, WqkvT, qkvB, qkvact,
                                                    nullptr, 3072, 1024);
  // 4) attention (100 windows x 16 heads)
  attn_kernel<<<dim3(16, 100), 256, 0, stream>>>(qkvact, rph, rpw, attnout);
  // 5) proj GEMM + unpartition + residual -> x2
  gemm_kernel<1><<<dim3(154, 8), 256, 0, stream>>>(attnout, WprojT, projB, x2,
                                                   x, 1024, 1024);
  // 6) LN2
  ln_kernel<false><<<dim3(16384), 256, 0, stream>>>(x2, n2s, n2b, z1);
  // 7) fc1 + GELU (16384 x 4096 x 1024)
  gemm_kernel<2><<<dim3(128, 32), 256, 0, stream>>>(z1, Wfc1T, fc1B, h1,
                                                    nullptr, 4096, 1024);
  // 8) fc2 + residual -> d_out (16384 x 1024 x 4096)
  gemm_kernel<3><<<dim3(128, 8), 256, 0, stream>>>(h1, Wfc2T, fc2B, d_out, x2,
                                                   1024, 4096);
}
